// SelfAttention_12747462934940
// MI455X (gfx1250) — matrix operand order repaired and checked by reading, not compiled, not run on hardware
//
#include <hip/hip_runtime.h>
#include <hip/hip_bf16.h>
#include <stdint.h>

// ---------------------------------------------------------------------------
// Self-attention for MI455X (gfx1250, wave32, WMMA).
// GEMMs: bf16 inputs, fp32 accumulate via v_wmma_f32_16x16x32_bf16.
// Staging: global_load_async_to_lds_b128 (ASYNCcnt) + LDS double buffering.
// ---------------------------------------------------------------------------

typedef __bf16 bf16_t;
typedef __attribute__((ext_vector_type(16))) __bf16 v16bf;
typedef __attribute__((ext_vector_type(8)))  __bf16 v8bf;
typedef __attribute__((ext_vector_type(8)))  float  v8f;

__device__ __forceinline__ bf16_t f2bf(float f) {
    union { float f; uint32_t u; } v; v.f = f;
    uint32_t r = v.u + 0x7FFFu + ((v.u >> 16) & 1u);   // round-to-nearest-even
    unsigned short h = (unsigned short)(r >> 16);
    bf16_t out;
    __builtin_memcpy(&out, &h, 2);
    return out;
}

__device__ __forceinline__ float bf2f(bf16_t b) {
    unsigned short h;
    __builtin_memcpy(&h, &b, 2);
    union { uint32_t u; float f; } v; v.u = ((uint32_t)h) << 16;
    return v.f;
}

// LDS byte offset of a shared-memory pointer: AMDGPU addrspacecast(AS3->flat)
// is {shared aperture base, 32-bit LDS offset}, so the low 32 bits are the
// offset the async-copy hardware wants (added to the LDS base and instruction offset).
__device__ __forceinline__ unsigned lds_off(const void* p) {
    return (unsigned)(uintptr_t)p;
}

// Issue one 16-byte async memory->LDS copy (tracked by ASYNCcnt).
__device__ __forceinline__ void async_copy_b128(unsigned ldsByteOff, const void* gptr) {
    asm volatile("global_load_async_to_lds_b128 %0, %1, off"
                 :: "v"(ldsByteOff), "v"((unsigned long long)(uintptr_t)gptr)
                 : "memory");
}

__device__ __forceinline__ void wait_async0() {
    asm volatile("s_wait_asynccnt 0x0" ::: "memory");
}

// ---------------------------------------------------------------------------
// fp32 -> bf16 elementwise conversion (grid-stride)
// ---------------------------------------------------------------------------
__global__ void __launch_bounds__(256) cvt_f32_bf16(const float* __restrict__ in,
                                                    bf16_t* __restrict__ out,
                                                    long long n) {
    long long i = (long long)blockIdx.x * blockDim.x + threadIdx.x;
    long long stride = (long long)gridDim.x * blockDim.x;
    for (; i < n; i += stride) out[i] = f2bf(in[i]);
}

// ---------------------------------------------------------------------------
// Tiled bf16 transpose: out[c][r] = in[r][c], per batch. 64x64 LDS tiles.
// Used to materialize K^T so the scores GEMM is a plain NN GEMM.
// ---------------------------------------------------------------------------
__global__ void __launch_bounds__(256) transpose_bf16(
    const bf16_t* __restrict__ in, bf16_t* __restrict__ out,
    int rows, int cols, long long inBatch, long long outBatch)
{
    __shared__ bf16_t tile[64][72];   // 72 = 64 + 8 pad; row = 144 B (16B-aligned)
    const bf16_t* inp  = in  + (long long)blockIdx.z * inBatch;
    bf16_t*       outp = out + (long long)blockIdx.z * outBatch;
    const int r0 = blockIdx.y * 64, c0 = blockIdx.x * 64;
    const int tid = threadIdx.x;

    #pragma unroll
    for (int i = 0; i < 2; ++i) {
        int c = tid + i * 256;                    // 512 chunks of 8 bf16
        int row = c >> 3, col = (c & 7) << 3;
        __builtin_memcpy(&tile[row][col],
                         &inp[(long long)(r0 + row) * cols + (c0 + col)], 16);
    }
    __syncthreads();
    #pragma unroll
    for (int i = 0; i < 2; ++i) {
        int c = tid + i * 256;
        int orow = c >> 3, ocol = (c & 7) << 3;   // out row = c0+orow, cols r0+ocol..+7
        bf16_t tmp[8];
        #pragma unroll
        for (int j = 0; j < 8; ++j) tmp[j] = tile[ocol + j][orow];
        __builtin_memcpy(&outp[(long long)(c0 + orow) * rows + (r0 + ocol)], tmp, 16);
    }
}

// ---------------------------------------------------------------------------
// NN GEMM, bf16 in / fp32 accumulate on v_wmma_f32_16x16x32_bf16.
//   C[b] = A[b] (MxK row-major) * B[b] (KxN row-major) (+ bias) (+ residual)
// Block: 256 threads (8 waves), BM=256, BN=128, BK=32.
// Wave grid 4x2; each wave computes a 64x64 region = 4x4 16x16 tiles
// -> 16 WMMAs per K-step per wave, 8 LDS fragment loads per step.
// Staging: async memory->LDS copies, double-buffered: issue K-tile k+1,
// compute on K-tile k, then s_wait_asynccnt 0 + one barrier per iteration.
// ---------------------------------------------------------------------------
template<bool OUTF32>
__global__ void __launch_bounds__(256) gemm_bf16_wmma(
    const bf16_t* __restrict__ A, const bf16_t* __restrict__ Bm,
    bf16_t* __restrict__ Cb, float* __restrict__ Cf,
    const float* __restrict__ bias, const float* __restrict__ residual,
    int M, int N, int K,
    long long aBatch, long long bBatch, long long cBatch)
{
    constexpr int LDA = 40;    // 32 + 8 pad (bf16); A row = 80 B (16B-aligned)
    constexpr int LDB = 136;   // 128 + 8 pad;      B row = 272 B (16B-aligned)
    __shared__ bf16_t As[2][256 * LDA];   // 2 x 20 KiB
    __shared__ bf16_t Bs[2][32 * LDB];    // 2 x 8.5 KiB

    const int batch = blockIdx.z;
    A  += (long long)batch * aBatch;
    Bm += (long long)batch * bBatch;
    const long long cOff = (long long)batch * cBatch;

    const int bm   = blockIdx.y * 256;
    const int bn   = blockIdx.x * 128;
    const int tid  = threadIdx.x;
    const int lane = tid & 31;
    const int wave = tid >> 5;
    const int waveM = wave >> 1;   // 0..3 (64 rows each)
    const int waveN = wave & 1;    // 0..1 (64 cols each)
    const int hh    = lane >> 4;   // lane half (ISA A/C layout)
    const int mrow  = lane & 15;

    v8f acc[4][4];
    #pragma unroll
    for (int i = 0; i < 4; ++i)
        #pragma unroll
        for (int j = 0; j < 4; ++j)
            acc[i][j] = (v8f){0.f, 0.f, 0.f, 0.f, 0.f, 0.f, 0.f, 0.f};

    // Issue all async copies for one K-tile into LDS buffer `buf` (no wait).
    auto stage = [&](int buf, int k0) {
        #pragma unroll
        for (int i = 0; i < 4; ++i) {              // A: 256x32 bf16 = 1024 chunks
            int c   = tid + i * 256;
            int row = c >> 2, col = (c & 3) << 3;
            async_copy_b128(lds_off(&As[buf][row * LDA + col]),
                            &A[(long long)(bm + row) * K + (k0 + col)]);
        }
        #pragma unroll
        for (int i = 0; i < 2; ++i) {              // B: 32x128 bf16 = 512 chunks
            int c   = tid + i * 256;
            int row = c >> 4, col = (c & 15) << 3;
            async_copy_b128(lds_off(&Bs[buf][row * LDB + col]),
                            &Bm[(long long)(k0 + row) * N + (bn + col)]);
        }
    };

    // Prologue: fill buffer 0.
    stage(0, 0);
    wait_async0();
    __syncthreads();

    int buf = 0;
    for (int k0 = 0; k0 < K; k0 += 32) {
        // Kick off the next tile while we compute on this one.
        if (k0 + 32 < K) stage(buf ^ 1, k0 + 32);

        // ---- B fragments: lane = N col (lane&15); element s = K row (table order, like A) ----
        v16bf bfrag[4];
        #pragma unroll
        for (int nt = 0; nt < 4; ++nt) {
            #pragma unroll
            for (int s = 0; s < 16; ++s)
                bfrag[nt][s] = Bs[buf][((s & 8) * 2 + hh * 8 + (s & 7)) * LDB + waveN * 64 + nt * 16 + mrow];
        }

        // ---- A fragments + 16 WMMAs ----
        #pragma unroll
        for (int mt = 0; mt < 4; ++mt) {
            const bf16_t* ap = &As[buf][(waveM * 64 + mt * 16 + mrow) * LDA];
            v8bf lo, hi;
            __builtin_memcpy(&lo, ap + hh * 8, 16);        // K = half*8 .. +7
            __builtin_memcpy(&hi, ap + 16 + hh * 8, 16);   // K = 16+half*8 .. +7
            v16bf afrag = __builtin_shufflevector(
                lo, hi, 0, 1, 2, 3, 4, 5, 6, 7, 8, 9, 10, 11, 12, 13, 14, 15);
            #pragma unroll
            for (int nt = 0; nt < 4; ++nt)
                acc[mt][nt] = __builtin_amdgcn_wmma_f32_16x16x32_bf16(
                    false, afrag, false, bfrag[nt],
                    (short)0, acc[mt][nt], false, false);
        }

        // All waves: my async copies landed; everyone done reading this buffer.
        wait_async0();
        __syncthreads();
        buf ^= 1;
    }

    // ---- Epilogue: C/D layout — VGPR r, lane l -> row r+(l>=16?8:0), col l&15 ----
    const int cN    = lane & 15;
    const int rBase = hh * 8;
    #pragma unroll
    for (int mt = 0; mt < 4; ++mt) {
        #pragma unroll
        for (int nt = 0; nt < 4; ++nt) {
            int gr = bm + waveM * 64 + mt * 16 + rBase;
            int gc = bn + waveN * 64 + nt * 16 + cN;
            float badd = bias ? bias[gc] : 0.f;
            #pragma unroll
            for (int r = 0; r < 8; ++r) {
                long long idx = cOff + (long long)(gr + r) * N + gc;
                float v = acc[mt][nt][r] + badd;
                if (residual) v += residual[idx];
                if (OUTF32) Cf[idx] = v;
                else        Cb[idx] = f2bf(v);
            }
        }
    }
}

// ---------------------------------------------------------------------------
// Row softmax over bf16 scores, in place. One 256-thread block per row of 2048.
// fp32 max/sum statistics; wave32 shuffle reduce + cross-wave LDS reduce.
// ---------------------------------------------------------------------------
__global__ void __launch_bounds__(256) softmax_rows(bf16_t* __restrict__ S, int ncols) {
    __shared__ float redMax[8];
    __shared__ float redSum[8];
    bf16_t* row = S + (long long)blockIdx.x * ncols;
    const int tid  = threadIdx.x;
    const int lane = tid & 31;
    const int wave = tid >> 5;

    float vals[8];
    float m = -1e30f;
    #pragma unroll
    for (int i = 0; i < 8; ++i) {
        vals[i] = bf2f(row[tid + i * 256]);
        m = fmaxf(m, vals[i]);
    }
    #pragma unroll
    for (int off = 16; off > 0; off >>= 1)
        m = fmaxf(m, __shfl_xor(m, off, 32));
    if (lane == 0) redMax[wave] = m;
    __syncthreads();
    m = redMax[0];
    #pragma unroll
    for (int w = 1; w < 8; ++w) m = fmaxf(m, redMax[w]);

    float s = 0.f;
    #pragma unroll
    for (int i = 0; i < 8; ++i) {
        vals[i] = __expf(vals[i] - m);
        s += vals[i];
    }
    #pragma unroll
    for (int off = 16; off > 0; off >>= 1)
        s += __shfl_xor(s, off, 32);
    if (lane == 0) redSum[wave] = s;
    __syncthreads();
    float tot = 0.f;
    #pragma unroll
    for (int w = 0; w < 8; ++w) tot += redSum[w];
    float inv = 1.0f / tot;

    #pragma unroll
    for (int i = 0; i < 8; ++i)
        row[tid + i * 256] = f2bf(vals[i] * inv);
}

// ---------------------------------------------------------------------------
// Host launcher
// ---------------------------------------------------------------------------
extern "C" void kernel_launch(void* const* d_in, const int* in_sizes, int n_in,
                              void* d_out, int out_size, void* d_ws, size_t ws_size,
                              hipStream_t stream) {
    (void)in_sizes; (void)n_in; (void)out_size; (void)ws_size;

    constexpr int  Bn = 4, Sn = 2048, Dn = 1024;
    constexpr long long BS = (long long)Bn * Sn;         // 8192 rows
    constexpr long long SD = (long long)Sn * Dn;         // per-batch QKV stride
    constexpr long long SS = (long long)Sn * Sn;         // per-batch score stride

    const float* x  = (const float*)d_in[0];
    const float* Wq = (const float*)d_in[1];
    const float* bq = (const float*)d_in[2];
    const float* Wk = (const float*)d_in[3];
    const float* bk = (const float*)d_in[4];
    const float* Wv = (const float*)d_in[5];
    const float* bv = (const float*)d_in[6];
    const float* Wo = (const float*)d_in[7];
    const float* bo = (const float*)d_in[8];
    float* out = (float*)d_out;

    // Workspace layout (bytes): total 136 MiB.
    char* ws = (char*)d_ws;
    bf16_t* xb   = (bf16_t*)(ws + (size_t)  0u * (1u << 20));  // 16 MiB: x bf16
    bf16_t* wqb  = (bf16_t*)(ws + (size_t) 16u * (1u << 20));  //  2 MiB
    bf16_t* wkb  = (bf16_t*)(ws + (size_t) 18u * (1u << 20));  //  2 MiB
    bf16_t* wvb  = (bf16_t*)(ws + (size_t) 20u * (1u << 20));  //  2 MiB
    bf16_t* wob  = (bf16_t*)(ws + (size_t) 22u * (1u << 20));  //  2 MiB
    bf16_t* Qb   = (bf16_t*)(ws + (size_t) 24u * (1u << 20));  // 16 MiB
    bf16_t* Kb   = (bf16_t*)(ws + (size_t) 40u * (1u << 20));  // 16 MiB
    bf16_t* Vb   = (bf16_t*)(ws + (size_t) 56u * (1u << 20));  // 16 MiB
    bf16_t* Pb   = (bf16_t*)(ws + (size_t) 72u * (1u << 20));  // 32 MiB: scores/probs
    bf16_t* Ctxb = (bf16_t*)(ws + (size_t)104u * (1u << 20));  // 16 MiB
    bf16_t* KTb  = (bf16_t*)(ws + (size_t)120u * (1u << 20));  // 16 MiB: K^T

    dim3 blk(256);

    // 1) fp32 -> bf16 conversions
    cvt_f32_bf16<<<2048, blk, 0, stream>>>(x,  xb,  BS * Dn);
    cvt_f32_bf16<<<1024, blk, 0, stream>>>(Wq, wqb, (long long)Dn * Dn);
    cvt_f32_bf16<<<1024, blk, 0, stream>>>(Wk, wkb, (long long)Dn * Dn);
    cvt_f32_bf16<<<1024, blk, 0, stream>>>(Wv, wvb, (long long)Dn * Dn);
    cvt_f32_bf16<<<1024, blk, 0, stream>>>(Wo, wob, (long long)Dn * Dn);

    // 2) Q/K/V projections: [8192,1024] x [1024,1024] + bias -> bf16
    dim3 gProj(Dn / 128, (unsigned)(BS / 256), 1);   // (8, 32, 1)
    gemm_bf16_wmma<false><<<gProj, blk, 0, stream>>>(
        xb, wqb, Qb, nullptr, bq, nullptr, (int)BS, Dn, Dn, 0, 0, 0);
    gemm_bf16_wmma<false><<<gProj, blk, 0, stream>>>(
        xb, wkb, Kb, nullptr, bk, nullptr, (int)BS, Dn, Dn, 0, 0, 0);
    gemm_bf16_wmma<false><<<gProj, blk, 0, stream>>>(
        xb, wvb, Vb, nullptr, bv, nullptr, (int)BS, Dn, Dn, 0, 0, 0);

    // 3) K^T per batch: [2048,1024] -> [1024,2048]
    dim3 gTr(Dn / 64, Sn / 64, Bn);                  // (16, 32, 4)
    transpose_bf16<<<gTr, blk, 0, stream>>>(Kb, KTb, Sn, Dn, SD, SD);

    // 4) scores = Q x K^T per batch (NN): [2048,1024] x [1024,2048] -> [2048,2048]
    dim3 gScores(Sn / 128, Sn / 256, Bn);            // (16, 8, 4)
    gemm_bf16_wmma<false><<<gScores, blk, 0, stream>>>(
        Qb, KTb, Pb, nullptr, nullptr, nullptr, Sn, Sn, Dn, SD, SD, SS);

    // 5) row softmax in place on bf16 scores
    softmax_rows<<<(unsigned)BS, blk, 0, stream>>>(Pb, Sn);

    // 6) ctx = P x V per batch: [2048,2048] x [2048,1024] -> [2048,1024]
    dim3 gCtx(Dn / 128, Sn / 256, Bn);               // (8, 8, 4)
    gemm_bf16_wmma<false><<<gCtx, blk, 0, stream>>>(
        Pb, Vb, Ctxb, nullptr, nullptr, nullptr, Sn, Dn, Sn, SS, SD, SD);

    // 7) out = ctx x Wo + bo + x  (fp32 output, fused residual)
    gemm_bf16_wmma<true><<<gProj, blk, 0, stream>>>(
        Ctxb, wob, nullptr, out, bo, x, (int)BS, Dn, Dn, 0, 0, 0);
}
